// Model_39676907881959
// MI455X (gfx1250) — matrix operand order repaired and checked by reading, not compiled, not run on hardware
//
#include <hip/hip_runtime.h>

// Flash-attention for: out = softmax(x1 @ x2^T / sqrt(32)) @ x2
// B=8, N=M=4096, D=64, fp32 in/out. f16 WMMA (f32 accum), async-to-LDS staging.

#define BATCH 8
#define NQ    4096
#define NK    4096
#define DH    64
#define BQ    128   // query rows per block (8 waves x 16)
#define KT    64    // keys per tile

typedef __attribute__((ext_vector_type(16))) _Float16 v16h;
typedef __attribute__((ext_vector_type(8)))  _Float16 v8h;
typedef __attribute__((ext_vector_type(4)))  _Float16 v4h;
typedef __attribute__((ext_vector_type(8)))  float    v8f;
typedef __attribute__((ext_vector_type(4)))  float    v4f;

static __device__ __forceinline__ v16h cat8(v8h lo, v8h hi) {
    return __builtin_shufflevector(lo, hi, 0,1,2,3,4,5,6,7,8,9,10,11,12,13,14,15);
}
static __device__ __forceinline__ v8f vzero8() {
    v8f z;
    #pragma unroll
    for (int i = 0; i < 8; ++i) z[i] = 0.0f;
    return z;
}

// ---- xor-butterfly within each 16-lane group -------------------------------
// v_permlane16_b32 keeps the reduction on the VALU pipe (the DS pipe is busy
// feeding WMMA B-fragments). Nibble i of {hi,lo} = source lane for lane i.
#if __has_builtin(__builtin_amdgcn_permlane16)
#define PERMX(v, lo, hi)                                                      \
    __uint_as_float(__builtin_amdgcn_permlane16(                              \
        __float_as_uint(v), __float_as_uint(v), (lo), (hi), false, false))
static __device__ __forceinline__ float redmax16(float t) {
    t = fmaxf(t, PERMX(t, 0x67452301u, 0xEFCDAB89u));   // xor 1
    t = fmaxf(t, PERMX(t, 0x54761032u, 0xDCFE98BAu));   // xor 2
    t = fmaxf(t, PERMX(t, 0x32107654u, 0xBA98FEDCu));   // xor 4
    t = fmaxf(t, PERMX(t, 0xFEDCBA98u, 0x76543210u));   // xor 8
    return t;
}
static __device__ __forceinline__ float redsum16(float t) {
    t += PERMX(t, 0x67452301u, 0xEFCDAB89u);
    t += PERMX(t, 0x54761032u, 0xDCFE98BAu);
    t += PERMX(t, 0x32107654u, 0xBA98FEDCu);
    t += PERMX(t, 0xFEDCBA98u, 0x76543210u);
    return t;
}
#else
static __device__ __forceinline__ float redmax16(float t) {
    t = fmaxf(t, __shfl_xor(t, 1, 32));
    t = fmaxf(t, __shfl_xor(t, 2, 32));
    t = fmaxf(t, __shfl_xor(t, 4, 32));
    t = fmaxf(t, __shfl_xor(t, 8, 32));
    return t;
}
static __device__ __forceinline__ float redsum16(float t) {
    t += __shfl_xor(t, 1, 32);
    t += __shfl_xor(t, 2, 32);
    t += __shfl_xor(t, 4, 32);
    t += __shfl_xor(t, 8, 32);
    return t;
}
#endif

// ---- async DMA of one contiguous 16KB x2 tile into LDS ---------------------
// Each lane copies 16B x 4; IOFFSET applies to both LDS and global address.
// Tracked by ASYNCcnt; completion enforced with s_wait_asynccnt + barrier.
static __device__ __forceinline__ void async_copy_tile(unsigned lds_off,
                                                       unsigned long long gaddr) {
    asm volatile(
        "global_load_async_to_lds_b128 %0, %1, off\n\t"
        "global_load_async_to_lds_b128 %0, %1, off offset:4096\n\t"
        "global_load_async_to_lds_b128 %0, %1, off offset:8192\n\t"
        "global_load_async_to_lds_b128 %0, %1, off offset:12288"
        :: "v"(lds_off), "v"(gaddr)
        : "memory");
}
static __device__ __forceinline__ void wait_async0() {
    asm volatile("s_wait_asynccnt 0x0" ::: "memory");
}

__global__ __launch_bounds__(256) void fa_fwd_kernel(const float* __restrict__ x1,
                                                     const float* __restrict__ x2,
                                                     float* __restrict__ out) {
    // Double-buffered raw f32 async landing zone + f16 dual-layout tiles +
    // per-wave P scratch (C-layout -> A-layout transpose).
    __shared__ alignas(16) float     sX [2][KT][DH];  // async DMA dest (f32)
    __shared__ alignas(16) _Float16 sK [KT][72];      // sK [key][d]
    __shared__ alignas(16) _Float16 sKt[DH][72];      // sKt[d][key]
    __shared__ alignas(16) _Float16 sP [8][16][72];   // per-wave P tile

    const int b     = blockIdx.y;
    const int qbase = blockIdx.x * BQ;
    const int tid   = threadIdx.x;
    const int wave  = tid >> 5;
    const int lane  = tid & 31;
    const int ln    = lane & 15;       // column / row-in-group id
    const int half  = lane >> 4;       // which 16-lane half
    const int abase = half * 8;        // A-matrix K-offset for this half

    const float scale = 0.17677669529663688f;   // 1/sqrt(32)
    const float LOG2E = 1.4426950408889634f;

    const unsigned long long x2tile0 =
        (unsigned long long)(uintptr_t)x2 + ((size_t)b * NK) * (DH * 4ull);

    // Prologue: kick off DMA for tile 0 while we load Q.
    async_copy_tile((unsigned)(uintptr_t)&sX[0][0][0] + (unsigned)tid * 16u,
                    x2tile0 + (unsigned)tid * 16u);

    // ---- Load Q fragments (A-matrix 16x32 f16 layout), pre-scaled ----
    // lane holds row m = ln; VGPRs 0-3: K = abase+0..7, VGPRs 4-7: K = 16+abase+0..7
    const int    qrow = qbase + wave * 16 + ln;
    const float* qptr = x1 + ((size_t)b * NQ + qrow) * DH;
    v16h Aq[2];
    #pragma unroll
    for (int f = 0; f < 2; ++f) {
        const v4f* p0 = (const v4f*)(qptr + f * 32 + abase);
        const v4f* p1 = (const v4f*)(qptr + f * 32 + 16 + abase);
        v4f a0 = __builtin_nontemporal_load(p0);
        v4f a1 = __builtin_nontemporal_load(p0 + 1);
        v4f b0 = __builtin_nontemporal_load(p1);
        v4f b1 = __builtin_nontemporal_load(p1 + 1);
        #pragma unroll
        for (int e = 0; e < 4; ++e) {
            Aq[f][e]      = (_Float16)(a0[e] * scale);
            Aq[f][4 + e]  = (_Float16)(a1[e] * scale);
            Aq[f][8 + e]  = (_Float16)(b0[e] * scale);
            Aq[f][12 + e] = (_Float16)(b1[e] * scale);
        }
    }

    // ---- Online softmax state: lane's half owns rows (half*8 + r), r=0..7 ----
    float mrow[8], lrow[8];
    v8f   O[4];
    #pragma unroll
    for (int r = 0; r < 8; ++r) { mrow[r] = -INFINITY; lrow[r] = 0.0f; }
    #pragma unroll
    for (int bd = 0; bd < 4; ++bd) O[bd] = vzero8();

    int cur = 0;
    for (int kt = 0; kt < NK; kt += KT) {
        // Tile `cur` DMA done (wave-local), barrier makes all waves' pieces
        // visible AND closes out last iteration's reads of sK/sKt.
        wait_async0();
        __syncthreads();

        // ---- Convert staged f32 tile -> f16 dual layout (row + transposed) ----
        {
            const int kk   = tid >> 2;
            const int dseg = (tid & 3) * 16;
            const v4f* src4 = (const v4f*)&sX[cur][kk][dseg];
            #pragma unroll
            for (int s4 = 0; s4 < 4; ++s4) {
                v4f v = src4[s4];
                _Float16 h0 = (_Float16)v[0], h1 = (_Float16)v[1];
                _Float16 h2 = (_Float16)v[2], h3 = (_Float16)v[3];
                v4h pk = {h0, h1, h2, h3};
                *(v4h*)&sK[kk][dseg + 4 * s4] = pk;
                sKt[dseg + 4 * s4 + 0][kk] = h0;
                sKt[dseg + 4 * s4 + 1][kk] = h1;
                sKt[dseg + 4 * s4 + 2][kk] = h2;
                sKt[dseg + 4 * s4 + 3][kk] = h3;
            }
        }

        // Prefetch next tile into the other buffer (overlaps with compute).
        if (kt + KT < NK) {
            async_copy_tile(
                (unsigned)(uintptr_t)&sX[cur ^ 1][0][0] + (unsigned)tid * 16u,
                x2tile0 + (size_t)(kt + KT) * (DH * 4ull) + (unsigned)tid * 16u);
        }
        __syncthreads();

        // ---- S = Q K^T for 4 key blocks of 16 (C-layout f32) ----
        v8f S[4];
        #pragma unroll
        for (int j = 0; j < 4; ++j) {
            v8f c = vzero8();
            #pragma unroll
            for (int ch = 0; ch < 2; ++ch) {
                // B 32x16: lane col n=ln, K(d) = ch*32 + half*8 + 0..7, ch*32 + 16 + half*8 + 0..7
                v8h blo = *(const v8h*)&sK[j * 16 + ln][ch * 32 + half * 8];
                v8h bhi = *(const v8h*)&sK[j * 16 + ln][ch * 32 + 16 + half * 8];
                v16h Bf = cat8(blo, bhi);
                c = __builtin_amdgcn_wmma_f32_16x16x32_f16(
                        false, Aq[ch], false, Bf, (short)0, c, false, false);
            }
            S[j] = c;
        }

        // ---- Row max (reduce over 16 lanes of this half) and rescale factor ----
        float alpha[8], psum[8];
        #pragma unroll
        for (int r = 0; r < 8; ++r) {
            float t = fmaxf(fmaxf(S[0][r], S[1][r]), fmaxf(S[2][r], S[3][r]));
            t = redmax16(t);
            float mnew = fmaxf(mrow[r], t);
            alpha[r]  = exp2f((mrow[r] - mnew) * LOG2E);
            mrow[r]   = mnew;
            psum[r]   = 0.0f;
        }

        // ---- P = exp(S - m); write per-wave f16 tile; accumulate row sums ----
        #pragma unroll
        for (int j = 0; j < 4; ++j) {
            #pragma unroll
            for (int r = 0; r < 8; ++r) {
                float p = exp2f((S[j][r] - mrow[r]) * LOG2E);
                psum[r] += p;
                sP[wave][r + half * 8][j * 16 + ln] = (_Float16)p;
            }
        }
        #pragma unroll
        for (int r = 0; r < 8; ++r)
            lrow[r] = lrow[r] * alpha[r] + redsum16(psum[r]);

        // ---- Rescale accumulators (row state is lane-local by construction) ----
        #pragma unroll
        for (int bd = 0; bd < 4; ++bd)
            #pragma unroll
            for (int r = 0; r < 8; ++r) O[bd][r] *= alpha[r];

        // ---- O += P V : A-frags of P from LDS, B-frags of V from transposed tile ----
        v16h Ap[2];
        #pragma unroll
        for (int kc = 0; kc < 2; ++kc) {
            v8h lo = *(const v8h*)&sP[wave][ln][kc * 32 + abase];
            v8h hi = *(const v8h*)&sP[wave][ln][kc * 32 + 16 + abase];
            Ap[kc] = cat8(lo, hi);
        }
        #pragma unroll
        for (int bd = 0; bd < 4; ++bd) {
            v8f c = O[bd];
            #pragma unroll
            for (int kc = 0; kc < 2; ++kc) {
                // B 32x16: lane col n=ln maps to d = bd*16+ln; K(key) = kc*32 + half*8 + 0..7, +16
                v8h blo = *(const v8h*)&sKt[bd * 16 + ln][kc * 32 + half * 8];
                v8h bhi = *(const v8h*)&sKt[bd * 16 + ln][kc * 32 + 16 + half * 8];
                v16h Bf = cat8(blo, bhi);
                c = __builtin_amdgcn_wmma_f32_16x16x32_f16(
                        false, Ap[kc], false, Bf, (short)0, c, false, false);
            }
            O[bd] = c;
        }
        cur ^= 1;
    }

    // ---- Normalize and store (f32, streaming) ----
    float inv[8];
    #pragma unroll
    for (int r = 0; r < 8; ++r) inv[r] = 1.0f / lrow[r];

    float* optr = out + ((size_t)b * NQ + qbase + wave * 16) * DH;
    #pragma unroll
    for (int bd = 0; bd < 4; ++bd) {
        #pragma unroll
        for (int r = 0; r < 8; ++r) {
            int row = r + half * 8;
            __builtin_nontemporal_store(O[bd][r] * inv[r],
                                        optr + row * DH + bd * 16 + ln);
        }
    }
}

extern "C" void kernel_launch(void* const* d_in, const int* in_sizes, int n_in,
                              void* d_out, int out_size, void* d_ws, size_t ws_size,
                              hipStream_t stream) {
    const float* x1  = (const float*)d_in[0];
    const float* x2  = (const float*)d_in[1];
    // d_in[2] = dropout_p (== 0) -> identity, ignored.
    float* out = (float*)d_out;

    dim3 grid(NQ / BQ, BATCH);   // (32, 8)
    fa_fwd_kernel<<<grid, 256, 0, stream>>>(x1, x2, out);
}
